// TemporalGCN_68109591380567
// MI455X (gfx1250) — matrix operand order repaired and checked by reading, not compiled, not run on hardware
//
#include <hip/hip_runtime.h>
#include <hip/hip_bf16.h>

typedef __attribute__((ext_vector_type(16))) _Float16 v16h;
typedef __attribute__((ext_vector_type(8)))  float    v8f;

#define B_   256
#define T_   2048
#define FIN  9
#define C1   16
#define C2   32
#define T1   1024
#define TP_  512
#define NNODE (B_*TP_)      // 131072
#define EEDGE 2097152
#define HID_ 64
#define OUT_ 6

// ---------------- conv1 (9->16, k=5, pad=2) + relu + maxpool2 ----------------
__global__ void conv1_relu_pool(const float* __restrict__ x,   // (B, T, 9)
                                const float* __restrict__ w,   // (16, 9, 5)
                                const float* __restrict__ bias,
                                float* __restrict__ h1)        // (B, 16, 1024)
{
    __shared__ float sw[C1 * FIN * 5];
    __shared__ float sb[C1];
    for (int i = threadIdx.x; i < C1 * FIN * 5; i += blockDim.x) sw[i] = w[i];
    if (threadIdx.x < C1) sb[threadIdx.x] = bias[threadIdx.x];
    __syncthreads();

    int tid = blockIdx.x * blockDim.x + threadIdx.x;   // over B*T1
    int b = tid / T1, p = tid % T1;

    float xin[6][FIN];
    int t0 = 2 * p - 2;
    #pragma unroll
    for (int s = 0; s < 6; ++s) {
        int t = t0 + s;
        bool ok = (t >= 0) && (t < T_);
        const float* xp = x + ((size_t)b * T_ + (ok ? t : 0)) * FIN;
        #pragma unroll
        for (int f = 0; f < FIN; ++f) xin[s][f] = ok ? xp[f] : 0.0f;
    }
    #pragma unroll
    for (int c = 0; c < C1; ++c) {
        float a0 = sb[c], a1 = sb[c];
        const float* wc = &sw[c * FIN * 5];
        #pragma unroll
        for (int f = 0; f < FIN; ++f)
            #pragma unroll
            for (int k = 0; k < 5; ++k) {
                float wv = wc[f * 5 + k];
                a0 = fmaf(xin[k][f],     wv, a0);   // time 2p   + k - 2
                a1 = fmaf(xin[k + 1][f], wv, a1);   // time 2p+1 + k - 2
            }
        h1[((size_t)b * C1 + c) * T1 + p] = fmaxf(fmaxf(a0, a1), 0.0f);
    }
}

// ---------------- conv2 (16->32, k=5, pad=2) + relu + maxpool2 ---------------
__global__ void conv2_relu_pool(const float* __restrict__ h1,  // (B, 16, 1024)
                                const float* __restrict__ w,   // (32, 16, 5)
                                const float* __restrict__ bias,
                                float* __restrict__ feat)      // (N, 32) node-major
{
    __shared__ float sw[C2 * C1 * 5];
    __shared__ float sb[C2];
    for (int i = threadIdx.x; i < C2 * C1 * 5; i += blockDim.x) sw[i] = w[i];
    if (threadIdx.x < C2) sb[threadIdx.x] = bias[threadIdx.x];
    __syncthreads();

    int tid = blockIdx.x * blockDim.x + threadIdx.x;   // over N
    int b = tid / TP_, tp = tid % TP_;

    float hin[C1][6];
    int p0 = 2 * tp - 2;
    #pragma unroll
    for (int ci = 0; ci < C1; ++ci) {
        const float* hp = h1 + ((size_t)b * C1 + ci) * T1;
        #pragma unroll
        for (int s = 0; s < 6; ++s) {
            int p = p0 + s;
            hin[ci][s] = (p >= 0 && p < T1) ? hp[p] : 0.0f;
        }
    }
    float* outp = feat + (size_t)tid * C2;
    #pragma unroll
    for (int c = 0; c < C2; ++c) {
        float a0 = sb[c], a1 = sb[c];
        const float* wc = &sw[c * C1 * 5];
        #pragma unroll
        for (int ci = 0; ci < C1; ++ci)
            #pragma unroll
            for (int k = 0; k < 5; ++k) {
                float wv = wc[ci * 5 + k];
                a0 = fmaf(hin[ci][k],     wv, a0);
                a1 = fmaf(hin[ci][k + 1], wv, a1);
            }
        outp[c] = fmaxf(fmaxf(a0, a1), 0.0f);
    }
}

// ---------------- degree (self loop = 1) + edge counts + rsqrt ---------------
__global__ void deg_init(float* __restrict__ deg) {
    int n = blockIdx.x * blockDim.x + threadIdx.x;
    deg[n] = 1.0f;
}
__global__ void deg_edges(const int* __restrict__ cols, float* __restrict__ deg) {
    int e = blockIdx.x * blockDim.x + threadIdx.x;
    atomicAdd(&deg[cols[e]], 1.0f);
}
__global__ void deg_rsqrt(float* __restrict__ deg) {
    int n = blockIdx.x * blockDim.x + threadIdx.x;
    deg[n] = rsqrtf(deg[n]);
}

// ---------------- WMMA GEMM: Dst(N,64) = [relu](A(N,K)) @ W(K,64) ------------
// Block = 128 threads = 4 waves, 64 output rows per block.
// Staging threads write f16 values directly into FRAGMENT order in LDS, so
// the compute loop reads each 16xf16 lane fragment as one aligned v16h load
// (2x ds_load_b128) and feeds v_wmma_f32_16x16x32_f16 directly.
//
// Fragment layouts (ISA 7.12.2, wave32):
//   A 16x32 f16 : lane = half*16 + m ; element j <-> kl bits
//                 [4]=j>>3, [3]=half, [2:1]=(j>>1)&3, [0]=j&1
//   B 32x16 f16 : lane = half*16 + n ; kl = (j>>3)*16 + half*8 + (j&7)
//   C/D 16x16 f32: vgpr r -> M = half*8 + r, N = lane&15
template <int K, bool RELU>
__global__ void gemm_wmma(const float* __restrict__ A,
                          const float* __restrict__ W,
                          float* __restrict__ Dst)
{
    constexpr int KC = K / 32;                       // k-chunks of 32
    __shared__ _Float16 fragA[4 * KC * 32 * 16];     // [wave][kc][lane][j]
    __shared__ _Float16 fragB[4 * KC * 32 * 16];     // [ct]  [kc][lane][j]

    const int tid = threadIdx.x;
    const int rowBlk = blockIdx.x * 64;

    // ---- stage A tile (64 x K) into fragment order ----
    for (int idx = tid; idx < 64 * K; idx += 128) {
        int r = idx / K, k = idx - r * K;
        float v = A[(size_t)(rowBlk + r) * K + k];
        if (RELU) v = fmaxf(v, 0.0f);
        int wave = r >> 4, m = r & 15;
        int kc = k >> 5, kl = k & 31;
        int half = (kl >> 3) & 1;
        int j = ((kl >> 4) << 3) | (((kl >> 1) & 3) << 1) | (kl & 1);
        int lane = (half << 4) | m;
        fragA[((wave * KC + kc) * 32 + lane) * 16 + j] = (_Float16)v;
    }
    // ---- stage W tile (K x 64) into fragment order ----
    for (int idx = tid; idx < K * 64; idx += 128) {
        int k = idx >> 6, n64 = idx & 63;
        int ct = n64 >> 4, n = n64 & 15;
        int kc = k >> 5, kl = k & 31;
        int half = (kl >> 3) & 1;
        int j = ((kl >> 4) << 3) | (kl & 7);
        int lane = (half << 4) | n;
        fragB[((ct * KC + kc) * 32 + lane) * 16 + j] = (_Float16)W[idx];
    }
    __syncthreads();

    const int lane = tid & 31;
    const int wave = tid >> 5;
    const int mn   = lane & 15;
    const int half = lane >> 4;

    v8f acc[4];
    #pragma unroll
    for (int ct = 0; ct < 4; ++ct)
        acc[ct] = (v8f){0.f,0.f,0.f,0.f,0.f,0.f,0.f,0.f};

    #pragma unroll
    for (int kc = 0; kc < KC; ++kc) {
        v16h a = *(const v16h*)&fragA[((wave * KC + kc) * 32 + lane) * 16];
        #pragma unroll
        for (int ct = 0; ct < 4; ++ct) {
            v16h bfr = *(const v16h*)&fragB[((ct * KC + kc) * 32 + lane) * 16];
            acc[ct] = __builtin_amdgcn_wmma_f32_16x16x32_f16(
                false, a, false, bfr, (short)0, acc[ct], false, false);
        }
    }

    #pragma unroll
    for (int ct = 0; ct < 4; ++ct) {
        #pragma unroll
        for (int r = 0; r < 8; ++r) {
            int row = rowBlk + wave * 16 + (half << 3) + r;
            Dst[(size_t)row * 64 + ct * 16 + mn] = acc[ct][r];
        }
    }
}

// ------------- scatter init: dst = bias + dinv[n]^2 * src (self loop) --------
__global__ void scatter_init(float* __restrict__ dst, const float* __restrict__ src,
                             const float* __restrict__ bias, const float* __restrict__ dinv)
{
    int tid = blockIdx.x * blockDim.x + threadIdx.x;   // N*64
    int n = tid >> 6, f = tid & 63;
    float dv = dinv[n];
    dst[tid] = bias[f] + dv * dv * src[tid];
}

// ------------- edge scatter: dst[col] += norm * src[row], 16 lanes/edge ------
__global__ void scatter_edges(float* __restrict__ dst, const float* __restrict__ src,
                              const int* __restrict__ rows, const int* __restrict__ cols,
                              const float* __restrict__ dinv)
{
    int tid = blockIdx.x * blockDim.x + threadIdx.x;   // E*16
    int e = tid >> 4, g = tid & 15;
    int r = rows[e], c = cols[e];
    float norm = dinv[r] * dinv[c];
    const float4 v = *(const float4*)(src + (size_t)r * 64 + g * 4);
    float* d = dst + (size_t)c * 64 + g * 4;
    atomicAdd(d + 0, v.x * norm);
    atomicAdd(d + 1, v.y * norm);
    atomicAdd(d + 2, v.z * norm);
    atomicAdd(d + 3, v.w * norm);
}

// ------------- relu + mean over TP + FC (64 -> 6) ----------------------------
__global__ void meanpool_fc(const float* __restrict__ g2,   // (N, 64) pre-relu
                            const float* __restrict__ fcw,  // (6, 64)
                            const float* __restrict__ fcb,
                            float* __restrict__ out)        // (B, 6)
{
    __shared__ float pooled[HID_];
    int b = blockIdx.x, f = threadIdx.x;   // 64 threads
    float s = 0.0f;
    const float* base = g2 + (size_t)b * TP_ * HID_ + f;
    for (int tp = 0; tp < TP_; ++tp) s += fmaxf(base[(size_t)tp * HID_], 0.0f);
    pooled[f] = s * (1.0f / TP_);
    __syncthreads();
    if (f < OUT_) {
        float o = fcb[f];
        #pragma unroll
        for (int k = 0; k < HID_; ++k) o = fmaf(pooled[k], fcw[f * HID_ + k], o);
        out[b * OUT_ + f] = o;
    }
}

extern "C" void kernel_launch(void* const* d_in, const int* in_sizes, int n_in,
                              void* d_out, int out_size, void* d_ws, size_t ws_size,
                              hipStream_t stream)
{
    (void)in_sizes; (void)n_in; (void)out_size; (void)ws_size;

    const float* x   = (const float*)d_in[0];
    const int*   eix = (const int*)  d_in[1];
    const float* c1w = (const float*)d_in[2];
    const float* c1b = (const float*)d_in[3];
    const float* c2w = (const float*)d_in[4];
    const float* c2b = (const float*)d_in[5];
    const float* g1w = (const float*)d_in[6];
    const float* g1b = (const float*)d_in[7];
    const float* g2w = (const float*)d_in[8];
    const float* g2b = (const float*)d_in[9];
    const float* fcw = (const float*)d_in[10];
    const float* fcb = (const float*)d_in[11];
    float* out = (float*)d_out;

    float* ws   = (float*)d_ws;
    float* h1   = ws;                          // B*16*1024        (16 MB)
    float* feat = h1   + (size_t)B_ * C1 * T1; // N*32             (16 MB)
    float* deg  = feat + (size_t)NNODE * C2;   // N                (0.5 MB)
    float* bufP = deg  + (size_t)NNODE;        // N*64  matmul out (33.5 MB)
    float* bufQ = bufP + (size_t)NNODE * HID_; // N*64  scatter acc(33.5 MB)

    const int* rows = eix;
    const int* cols = eix + EEDGE;

    // temporal conv stack
    conv1_relu_pool<<<(B_ * T1) / 256, 256, 0, stream>>>(x, c1w, c1b, h1);
    conv2_relu_pool<<<NNODE / 256, 256, 0, stream>>>(h1, c2w, c2b, feat);

    // symmetric normalization
    deg_init <<<NNODE / 256, 256, 0, stream>>>(deg);
    deg_edges<<<EEDGE / 256, 256, 0, stream>>>(cols, deg);
    deg_rsqrt<<<NNODE / 256, 256, 0, stream>>>(deg);

    // GCN layer 1: feat(N,32) @ W1(32,64) -> bufP ; scatter (+bias, self loop) -> bufQ
    gemm_wmma<32, false><<<NNODE / 64, 128, 0, stream>>>(feat, g1w, bufP);
    scatter_init<<<(NNODE * HID_) / 256, 256, 0, stream>>>(bufQ, bufP, g1b, deg);
    scatter_edges<<<((size_t)EEDGE * 16) / 256, 256, 0, stream>>>(bufQ, bufP, rows, cols, deg);

    // GCN layer 2: relu fused into A-load; ping-pong buffers
    gemm_wmma<64, true><<<NNODE / 64, 128, 0, stream>>>(bufQ, g2w, bufP);
    scatter_init<<<(NNODE * HID_) / 256, 256, 0, stream>>>(bufQ, bufP, g2b, deg);
    scatter_edges<<<((size_t)EEDGE * 16) / 256, 256, 0, stream>>>(bufQ, bufP, rows, cols, deg);

    // relu + temporal mean pool + FC
    meanpool_fc<<<B_, HID_, 0, stream>>>(bufQ, fcw, fcb, out);
}
